// ModernNet_3212635537425
// MI455X (gfx1250) — matrix operand order repaired and checked by reading, not compiled, not run on hardware
//
#include <hip/hip_runtime.h>

typedef __attribute__((ext_vector_type(16))) _Float16 v16h;
typedef __attribute__((ext_vector_type(8)))  _Float16 v8h;
typedef __attribute__((ext_vector_type(2)))  _Float16 v2h;
typedef __attribute__((ext_vector_type(8)))  float    v8f;

#define NWAVES 2
#define BLOCK  (NWAVES * 32)

// ---------------- LDS layout (in f16 halves unless noted) ----------------
// per-wave region0: padded input images 16 x (20x20 + 2 slack) = 6432, + zero slot.
//   later reused as act2 (16 x 200, 192 used) + act3 (16 x 40, 32 used).
// per-wave act1 (padded): 16 x (12ch x (12x12 +1 slack) + 6 slack) = 16 x 1746.
// strides chosen so lane-varying strides are odd dword counts (bank spread),
// and act2/act3 strides are 16B x odd (conflict-free ds_load_b128).
static constexpr int IMG_STRIDE  = 402;     // 804B -> 201 dwords (odd)
static constexpr int ZSLOT       = 6432;    // per-wave zero slot (4 halves)
static constexpr int R0_HALVES   = 6464;
static constexpr int ACT2_STRIDE = 200;     // 400B = 16B * 25 (odd)
static constexpr int ACT3_BASE   = 3200;
static constexpr int ACT3_STRIDE = 40;      // 80B = 16B * 5 (odd)
static constexpr int A1_CH       = 145;     // 12x12 + 1 slack
static constexpr int A1_IMG      = 1746;    // 3492B -> 873 dwords (odd)
static constexpr int A1_HALVES   = 16 * A1_IMG;               // 27936
static constexpr int A1_BASE_H   = NWAVES * R0_HALVES;        // halves
static constexpr int BIAS_BASE_B = 2 * (NWAVES * R0_HALVES + NWAVES * A1_HALVES);
static constexpr int H1B_OFF = BIAS_BASE_B;                   // 768 f32
static constexpr int H2B_OFF = H1B_OFF + 3072;                // 192 f32
static constexpr int H3B_OFF = H2B_OFF + 768;                 // 32 f32 (zero padded)
static constexpr int OB_OFF  = H3B_OFF + 128;                 // 16 f32 (zero padded)
static constexpr int SMEM_BYTES = OB_OFF + 64;                // 141632 B -> 2 WGs / 320KB WGP

__device__ __forceinline__ v8f wmma_f16(v16h a, v16h b, v8f c) {
  return __builtin_amdgcn_wmma_f32_16x16x32_f16(false, a, false, b, (short)0, c, false, false);
}
__device__ __forceinline__ void lds_fence() {
  asm volatile("s_wait_dscnt 0" ::: "memory");   // CDNA5 split dep-counter wait
}
// offset of conv tap k (0..24) inside a padded 20-wide input image row-major tile
__device__ __forceinline__ constexpr int d20(int k) { return (k / 5) * 20 + (k % 5); }
// offset of merged-K index k (0..299) inside padded act1: [ch][12x12(+1)]
__device__ __forceinline__ constexpr int cA1(int k) {
  int ch = k / 25, tp = k % 25;
  return ch * A1_CH + (tp / 5) * 12 + (tp % 5);
}

extern "C" __global__ __launch_bounds__(BLOCK) void
lenet1989_wmma(const float* __restrict__ x,
               const float* __restrict__ w1, const float* __restrict__ b1,
               const float* __restrict__ w2, const float* __restrict__ b2,
               const float* __restrict__ w3, const float* __restrict__ b3,
               const float* __restrict__ w4, const float* __restrict__ b4,
               float* __restrict__ out)
{
  extern __shared__ char smem[];
  _Float16* smemH = (_Float16*)smem;
  const int tid  = threadIdx.x;
  const int wave = tid >> 5;
  const int lane = tid & 31;
  const int L    = lane >> 4;         // half-wave selects operand K mapping
  const bool Lhi = (L != 0);
  const int n    = lane & 15;         // B/C/D column; also A row (image)
  const int nc   = (n < 12) ? n : 0;

  float* sH1b = (float*)(smem + H1B_OFF);
  float* sH2b = (float*)(smem + H2B_OFF);
  float* sH3b = (float*)(smem + H3B_OFF);
  float* sOb  = (float*)(smem + OB_OFF);
  for (int e = tid; e < 768; e += BLOCK) sH1b[e] = b1[e];
  for (int e = tid; e < 192; e += BLOCK) sH2b[e] = b2[e];
  for (int e = tid; e < 32;  e += BLOCK) sH3b[e] = (e < 30) ? b3[e] : 0.0f;
  for (int e = tid; e < 16;  e += BLOCK) sOb[e]  = (e < 10) ? b4[e] : 0.0f;

  const int r0 = wave * R0_HALVES;
  const int a1 = A1_BASE_H + wave * A1_HALVES;
  const int zslotAbs = r0 + ZSLOT;

  // ---- wave-local: fill padded regions with -1 (becomes the -1 padding) ----
  {
    v8h neg1;
#pragma unroll
    for (int i = 0; i < 8; i++) neg1[i] = (_Float16)(-1.0f);
    for (int i = lane; i < R0_HALVES / 8; i += 32) *(v8h*)(smemH + r0 + i * 8) = neg1;
    for (int i = lane; i < A1_HALVES / 8; i += 32) *(v8h*)(smemH + a1 + i * 8) = neg1;
  }
  // ---- wave-local: stage 16 images into padded 20x20 f16 tiles ----
  const long imgBase = ((long)blockIdx.x * NWAVES + wave) * 16;
  const float4* gin4 = (const float4*)(x + imgBase * 256);
#pragma unroll 2
  for (int t = 0; t < 32; t++) {
    int fi = t * 32 + lane;                 // 1024 float4 chunks
    float4 v = gin4[fi];
    int img = fi >> 6, rem = fi & 63, y = rem >> 2, xq = rem & 3;
    int d = r0 + img * IMG_STRIDE + (y + 2) * 20 + (4 * xq + 2);
    v2h lo = { (_Float16)v.x, (_Float16)v.y };
    v2h hi = { (_Float16)v.z, (_Float16)v.w };
    *(v2h*)(smemH + d)     = lo;
    *(v2h*)(smemH + d + 2) = hi;
  }
  if (lane < 4) smemH[zslotAbs + lane] = (_Float16)0.0f;   // K-pad zero slot
  __syncthreads();                                          // bias tables ready

  // ---- B1 (25x12 -> 32x16) built per-wave into registers (mask-multiply) ----
  v16h B1r;
#pragma unroll
  for (int h = 0; h < 16; h++) {
    int k = h + 16 * L;                       // B layout: col per lane, K = h + 16*L
    bool kv = Lhi ? (h < 9) : true;           // k < 25
    float m = (n < 12 && kv) ? 1.0f : 0.0f;
    int ka = kv ? k : 0;
    B1r[h] = (_Float16)(w1[nc * 25 + ka] * m);
  }

  // ================= Layer 1: 64 px, Y[16,12] = patch[16,32] x W1 =================
  lds_fence();
  const int myImg = r0 + n * IMG_STRIDE;
  const int myA1c = a1 + nc * A1_CH;
#pragma unroll 1
  for (int p = 0; p < 64; p++) {
    int oy = p >> 3, ox = p & 7;
    int base = myImg + 40 * oy + 2 * ox;      // padded coords: (2oy)*20 + 2ox
    v16h A;
#pragma unroll
    for (int h = 0; h < 16; h++) {            // A layout: k = h+16L
      int k0 = h;
      int off;
      if (h < 9) {                            // both variants valid (k<25)
        off = base + (Lhi ? d20(k0 + 16) : d20(k0));
      } else {                                // L=1 variant is K padding -> zero slot
        off = Lhi ? zslotAbs : (base + d20(k0));
      }
      A[h] = smemH[off];                      // unconditional, batched ds_load_u16
    }
    float bias = sH1b[nc * 64 + p];
    v8f acc;
#pragma unroll
    for (int r = 0; r < 8; r++) acc[r] = bias;
    acc = wmma_f16(A, B1r, acc);
    int sb = myA1c + (oy + 2) * 12 + (ox + 2);
#pragma unroll
    for (int r = 0; r < 8; r++) {
      if (n < 12) {
        int m = r + 8 * L;                    // D row layout
        smemH[sb + m * A1_IMG] = (_Float16)fmaxf(acc[r], 0.0f);
      }
    }
  }

  // ---- B2: merged 3-group weights (300x12 -> 10 x 32x16) into registers ----
  v16h B2r[10];
  {
    int g = n >> 2;
#pragma unroll
    for (int t = 0; t < 10; t++) {
#pragma unroll
      for (int h = 0; h < 16; h++) {
        int gk = 32 * t + h + 16 * L;
        bool kv = gk < 300;
        int kc = kv ? gk : 0;
        int ch = kc / 25, tap = kc % 25;
        bool mem = (g == 0) ? (ch < 8)
                 : (g == 1) ? (ch >= 4)
                            : ((ch < 4) || (ch >= 8));
        int cp  = (g == 0) ? ch : (g == 1) ? (ch - 4) : ((ch < 4) ? ch : ch - 4);
        int cpc = (cp >= 0 && cp < 8) ? cp : 0;
        float m = (n < 12 && kv && mem) ? 1.0f : 0.0f;
        B2r[t][h] = (_Float16)(w2[(nc * 8 + cpc) * 25 + tap] * m);
      }
    }
  }

  // ================= Layer 2: 16 px, Y[16,12] = im2col[16,320] x W2merged =================
  lds_fence();
  const int myPA = a1 + n * A1_IMG;           // gather base: image n, padded act1
  _Float16* act2H = smemH + r0;               // region0 reuse (input dead)
#pragma unroll 1
  for (int q = 0; q < 16; q++) {
    int oy = q >> 2, ox = q & 3;
    int qb = myPA + 24 * oy + 2 * ox;         // (2oy)*12 + 2ox
    float bias = sH2b[nc * 16 + q];
    v8f acc;
#pragma unroll
    for (int r = 0; r < 8; r++) acc[r] = bias;
#pragma unroll
    for (int t = 0; t < 10; t++) {
      v16h A;
#pragma unroll
      for (int h = 0; h < 16; h++) {
        int k0 = 32 * t + h;
        int k1 = k0 + 16;
        if (k1 < 300) {                       // both variants valid (tile t<9)
          A[h] = smemH[qb + (Lhi ? cA1(k1) : cA1(k0))];
        } else if (k0 < 300) {                // t=9, h=0..11: L=1 is K padding
          int off = Lhi ? zslotAbs : (qb + cA1(k0));
          A[h] = smemH[off];
        } else {                              // t=9, h>=12: both variants K padding
          A[h] = (_Float16)0.0f;
        }
      }
      acc = wmma_f16(A, B2r[t], acc);
    }
#pragma unroll
    for (int r = 0; r < 8; r++) {
      if (n < 12) {
        int m = r + 8 * L;
        act2H[m * ACT2_STRIDE + n * 16 + q] = (_Float16)fmaxf(acc[r], 0.0f);
      }
    }
  }

  // ---- B3 (192x30 -> 12 x 32x16) into registers ----
  v16h B3r[12];
#pragma unroll
  for (int nt = 0; nt < 2; nt++) {
#pragma unroll
    for (int kt = 0; kt < 6; kt++) {
#pragma unroll
      for (int h = 0; h < 16; h++) {
        int gk = 32 * kt + h + 16 * L;        // always < 192
        int col = nt * 16 + n;
        float m = (col < 30) ? 1.0f : 0.0f;
        int colc = (col < 30) ? col : 0;
        B3r[nt * 6 + kt][h] = (_Float16)(w3[gk * 30 + colc] * m);
      }
    }
  }

  // ================= Layer 3: Y[16,30] = act2[16,192] x W3 =================
  lds_fence();
  const _Float16* myAct2 = act2H + n * ACT2_STRIDE;
  _Float16* act3H = smemH + r0 + ACT3_BASE;
  v8f acc0, acc1;
  {
    float bb0 = sH3b[n], bb1 = sH3b[16 + n];
#pragma unroll
    for (int r = 0; r < 8; r++) { acc0[r] = bb0; acc1[r] = bb1; }
  }
#pragma unroll
  for (int t = 0; t < 6; t++) {
    v8h lo = *(const v8h*)(myAct2 + 32 * t + 16 * L);      // k = 16L..16L+7
    v8h hi = *(const v8h*)(myAct2 + 32 * t + 16 * L + 8);  // k = 16L+8..16L+15
    v16h A = __builtin_shufflevector(lo, hi, 0,1,2,3,4,5,6,7,8,9,10,11,12,13,14,15);
    acc0 = wmma_f16(A, B3r[t],     acc0);
    acc1 = wmma_f16(A, B3r[6 + t], acc1);
  }
#pragma unroll
  for (int r = 0; r < 8; r++) {
    int m = r + 8 * L;
    act3H[m * ACT3_STRIDE + n] = (_Float16)fmaxf(acc0[r], 0.0f);
    float v1 = (n < 14) ? fmaxf(acc1[r], 0.0f) : 0.0f;     // col 16+n < 30
    act3H[m * ACT3_STRIDE + 16 + n] = (_Float16)v1;
  }

  // ================= Layer 4: Y[16,10] = act3[16,32] x W4 =================
  v16h B4r;
#pragma unroll
  for (int h = 0; h < 16; h++) {
    int k = h + 16 * L;
    bool kv = Lhi ? (h < 14) : true;          // k < 30
    float m = (kv && n < 10) ? 1.0f : 0.0f;
    int kcl = kv ? k : 0;
    int ncl = (n < 10) ? n : 0;
    B4r[h] = (_Float16)(w4[kcl * 10 + ncl] * m);
  }
  lds_fence();
  {
    const _Float16* myA3 = act3H + n * ACT3_STRIDE;
    v8h lo = *(const v8h*)(myA3 + 16 * L);
    v8h hi = *(const v8h*)(myA3 + 16 * L + 8);
    v16h A = __builtin_shufflevector(lo, hi, 0,1,2,3,4,5,6,7,8,9,10,11,12,13,14,15);
    float bias = sOb[n];
    v8f acc;
#pragma unroll
    for (int r = 0; r < 8; r++) acc[r] = bias;
    acc = wmma_f16(A, B4r, acc);
#pragma unroll
    for (int r = 0; r < 8; r++) {
      if (n < 10) {
        long m = imgBase + r + 8 * L;
        out[m * 10 + n] = acc[r];
      }
    }
  }
}

extern "C" void kernel_launch(void* const* d_in, const int* in_sizes, int n_in,
                              void* d_out, int out_size, void* d_ws, size_t ws_size,
                              hipStream_t stream) {
  (void)n_in; (void)out_size; (void)d_ws; (void)ws_size;
  const float* x  = (const float*)d_in[0];
  const float* w1 = (const float*)d_in[1];
  const float* b1 = (const float*)d_in[2];
  const float* w2 = (const float*)d_in[3];
  const float* b2 = (const float*)d_in[4];
  const float* w3 = (const float*)d_in[5];
  const float* b3 = (const float*)d_in[6];
  const float* w4 = (const float*)d_in[7];
  const float* b4 = (const float*)d_in[8];
  float* out = (float*)d_out;

  const int Btot   = in_sizes[0] / 256;          // images (131072)
  const int blocks = Btot / (16 * NWAVES);       // 16 images per wave -> 4096 blocks

  (void)hipFuncSetAttribute((const void*)lenet1989_wmma,
                            hipFuncAttributeMaxDynamicSharedMemorySize, SMEM_BYTES);
  lenet1989_wmma<<<blocks, BLOCK, SMEM_BYTES, stream>>>(x, w1, b1, w2, b2, w3, b3, w4, b4, out);
}
